// NodeTaskHead_38019050504720
// MI455X (gfx1250) — compile-verified, matrix operand order checked by reading, not run on hardware
//
#include <hip/hip_runtime.h>
#include <hip/hip_bf16.h>
#include <stdint.h>

#define NC 4
#define NN 512
#define EE 512
#define PP 64
#define HH 32
#define DD 16
#define LN_EPS 1e-5f
#define SCALING 0.25f   // HEAD_DIM^-0.5 = 16^-0.5

typedef __bf16 bf16_t;
typedef __attribute__((ext_vector_type(16))) __bf16 v16bf;
typedef __attribute__((ext_vector_type(8)))  __bf16 v8bf;
typedef __attribute__((ext_vector_type(8)))  float  v8f;

static __device__ __forceinline__ bf16_t f2bf(float f) {
  return static_cast<bf16_t>(f);   // compiler picks hw cvt (RTNE)
}

// Fragment loader: A (16x32 MxK) or B (32x16 KxN read from [N][K] row-major),
// bf16 source, ld in elements. Per-lane k mapping per CDNA5 ISA 7.12.2.
static __device__ __forceinline__ v16bf load_frag_bf16(const bf16_t* p, int ld) {
  int lane = threadIdx.x & 31;
  int mn = lane & 15;
  int khalf = (lane >> 4) << 3;
  const bf16_t* r = p + mn * ld + khalf;
  v8bf lo = *(const v8bf*)(r);        // k = khalf .. khalf+7
  v8bf hi = *(const v8bf*)(r + 16);   // k = khalf+16 .. khalf+23
  return __builtin_shufflevector(lo, hi, 0,1,2,3,4,5,6,7,8,9,10,11,12,13,14,15);
}

// ---------------- LN(query) -> bf16 xq [2048, 512] ----------------
__global__ void k_ln_query(const float* __restrict__ qin, const float* __restrict__ g,
                           const float* __restrict__ b, bf16_t* __restrict__ xq) {
  int row = blockIdx.x;
  int t = threadIdx.x;                       // 256 threads
  const float* x = qin + (size_t)row * EE;
  float v0 = x[t], v1 = x[t + 256];
  float s = v0 + v1, sq = v0 * v0 + v1 * v1;
  #pragma unroll
  for (int o = 16; o >= 1; o >>= 1) { s += __shfl_xor(s, o, 32); sq += __shfl_xor(sq, o, 32); }
  __shared__ float ws[8][2];
  __shared__ float mv[2];
  int wave = t >> 5, lane = t & 31;
  if (lane == 0) { ws[wave][0] = s; ws[wave][1] = sq; }
  __syncthreads();
  if (t == 0) {
    float S = 0.f, Q = 0.f;
    #pragma unroll
    for (int w = 0; w < 8; ++w) { S += ws[w][0]; Q += ws[w][1]; }
    float mu = S * (1.0f / EE);
    mv[0] = mu;
    mv[1] = rsqrtf(Q * (1.0f / EE) - mu * mu + LN_EPS);
  }
  __syncthreads();
  float mu = mv[0], rs = mv[1];
  xq[(size_t)row * EE + t]       = f2bf((v0 - mu) * rs * g[t]       + b[t]);
  xq[(size_t)row * EE + t + 256] = f2bf((v1 - mu) * rs * g[t + 256] + b[t + 256]);
}

// ---------------- one-shot weight conversion to bf16 ----------------
// Wq, Wk: 512x512 ; Wb: 32x64 ; Wvf[h][e] = sum_d Wv[h*16+d][e]*Wf[h*16+d]
__global__ void k_prep_weights(const float* __restrict__ Wq, const float* __restrict__ Wk,
                               const float* __restrict__ Wv, const float* __restrict__ Wb,
                               const float* __restrict__ Wf,
                               bf16_t* __restrict__ Wq_bf, bf16_t* __restrict__ Wk_bf,
                               bf16_t* __restrict__ Wvf_bf, bf16_t* __restrict__ Wb_bf) {
  int idx = blockIdx.x * 256 + threadIdx.x;     // 0 .. 262143
  Wq_bf[idx] = f2bf(Wq[idx]);
  Wk_bf[idx] = f2bf(Wk[idx]);
  if (idx < HH * PP) Wb_bf[idx] = f2bf(Wb[idx]);
  if (idx < HH * EE) {
    int h = idx >> 9, e = idx & 511;
    float s = 0.f;
    #pragma unroll
    for (int d = 0; d < DD; ++d)
      s += Wv[(size_t)(h * DD + d) * EE + e] * Wf[h * DD + d];
    Wvf_bf[idx] = f2bf(s);
  }
}

// ---------------- q / kT / pvT GEMMs via WMMA bf16, 2x2 blocked ----------------
// sel 0: qbuf[row][o] = scaling * xq@Wq^T   (row-major [2048,512])
// sel 1: kT[c][o][j]  = xq@Wk^T transposed  ([4][512][512])
// sel 2: pvT[c][h][j] = xq@Wvf^T transposed ([4][32][512])
static __device__ __forceinline__ void store_tile(const v8f& acc, int sel, int rt, int ct,
                                                  float* __restrict__ qbuf,
                                                  float* __restrict__ kT,
                                                  float* __restrict__ pvT, int lane) {
  int nn = lane & 15;
  int mb = (lane >> 4) << 3;
  if (sel == 0) {
    int row = rt * 16 + mb, col = ct * 16 + nn;
    #pragma unroll
    for (int r = 0; r < 8; ++r) qbuf[(size_t)(row + r) * EE + col] = acc[r] * SCALING;
  } else if (sel == 1) {
    int c = rt >> 5, jb = (rt & 31) * 16 + mb, hd = ct * 16 + nn;
    float* base = kT + ((size_t)c * EE + hd) * NN + jb;
    #pragma unroll
    for (int r = 0; r < 8; ++r) base[r] = acc[r];
  } else {
    int c = rt >> 5, jb = (rt & 31) * 16 + mb, h = ct * 16 + nn;
    float* base = pvT + ((size_t)c * HH + h) * NN + jb;
    #pragma unroll
    for (int r = 0; r < 8; ++r) base[r] = acc[r];
  }
}

__global__ void k_qkv_gemm(const bf16_t* __restrict__ xq,
                           const bf16_t* __restrict__ Wq, const bf16_t* __restrict__ Wk,
                           const bf16_t* __restrict__ Wvf,
                           float* __restrict__ qbuf, float* __restrict__ kT,
                           float* __restrict__ pvT) {
  int wave = threadIdx.x >> 5;
  int lane = threadIdx.x & 31;
  int sel = blockIdx.y;
  int gw = blockIdx.x * 8 + wave;            // 2x2-tile job id
  int cg_n = (sel == 2) ? 1 : 16;            // col groups (2 tiles each)
  if (gw >= 64 * cg_n) return;               // 64 row groups (2 tiles each)
  int rg = gw / cg_n, cg = gw % cg_n;
  const bf16_t* W = (sel == 0) ? Wq : (sel == 1) ? Wk : Wvf;
  const bf16_t* A0 = xq + (size_t)rg * 32 * EE;
  const bf16_t* B0 = W + (size_t)cg * 32 * EE;
  v8f acc00 = {0.f,0.f,0.f,0.f,0.f,0.f,0.f,0.f};
  v8f acc01 = acc00, acc10 = acc00, acc11 = acc00;
  #pragma unroll 2
  for (int kk = 0; kk < 16; ++kk) {
    v16bf a0 = load_frag_bf16(A0 + kk * 32, EE);
    v16bf a1 = load_frag_bf16(A0 + (size_t)16 * EE + kk * 32, EE);
    v16bf b0 = load_frag_bf16(B0 + kk * 32, EE);
    v16bf b1 = load_frag_bf16(B0 + (size_t)16 * EE + kk * 32, EE);
    acc00 = __builtin_amdgcn_wmma_f32_16x16x32_bf16(false, a0, false, b0, (short)0, acc00, false, false);
    acc01 = __builtin_amdgcn_wmma_f32_16x16x32_bf16(false, a0, false, b1, (short)0, acc01, false, false);
    acc10 = __builtin_amdgcn_wmma_f32_16x16x32_bf16(false, a1, false, b0, (short)0, acc10, false, false);
    acc11 = __builtin_amdgcn_wmma_f32_16x16x32_bf16(false, a1, false, b1, (short)0, acc11, false, false);
  }
  int rt = rg * 2, ct = cg * 2;
  store_tile(acc00, sel, rt,     ct,     qbuf, kT, pvT, lane);
  store_tile(acc01, sel, rt,     ct + 1, qbuf, kT, pvT, lane);
  store_tile(acc10, sel, rt + 1, ct,     qbuf, kT, pvT, lane);
  store_tile(acc11, sel, rt + 1, ct + 1, qbuf, kT, pvT, lane);
}

// ---------------- fused attention per (c, i) ----------------
#define XP_LD 72    // bf16 stride (bank-conflict-free WMMA A loads)
#define AT_LD 514   // f32 stride  (bank-conflict-free column access)
#define ST_LD 68    // f32 staging stride (rows stay 16B aligned)

__global__ __launch_bounds__(512, 1)
void k_attn(const float* __restrict__ pair, const float* __restrict__ delta,
            const float* __restrict__ lpg, const float* __restrict__ lpb,
            const bf16_t* __restrict__ Wb_bf,
            const float* __restrict__ qbuf, const float* __restrict__ kT,
            const float* __restrict__ pvT, float* __restrict__ out) {
  __shared__ float  stage_s[2][64][ST_LD];  // async-staged raw pair rows (f32)
  __shared__ bf16_t xp_s[NN][XP_LD];        // LN(pair[c,i,:,:]) in bf16
  __shared__ float  at_s[HH][AT_LD];        // bias -> logits -> exp
  __shared__ float  qrow_s[EE];
  __shared__ float  inv_s[HH];
  __shared__ float  red_s[16][4];

  int tid = threadIdx.x;               // 512 threads = 16 waves
  int lane = tid & 31, wave = tid >> 5;
  int ci = blockIdx.x;
  int c = ci >> 9;

  qrow_s[tid] = qbuf[(size_t)ci * EE + tid];

  // prefetch this block's delta_pos row (consumed after softmax)
  const float* dp = delta + (size_t)ci * NN * 3 + tid * 3;
  __builtin_prefetch(dp, 0, 1);

  // --- async-pipelined LN(pair): 8 threads/row, 64 rows/pass, 8 passes ---
  // Each wave loads and consumes only its own 4 rows per pass, so per-wave
  // ASYNCcnt waits are sufficient (async loads retire in order).
  const float* prow = pair + (size_t)ci * NN * PP;
  int rr = tid >> 3;        // staging row 0..63
  int sub = tid & 7;        // 8 threads per row

  #define ISSUE_PASS(pass_, buf_)                                              \
    do {                                                                       \
      const float* gp_ = prow + (size_t)((pass_) * 64 + rr) * PP + sub * 8;    \
      uint32_t l0_ = (uint32_t)(uintptr_t)&stage_s[(buf_)][rr][sub * 8];       \
      uint64_t g0_ = (uint64_t)(uintptr_t)gp_;                                 \
      asm volatile("global_load_async_to_lds_b128 %0, %1, off"                 \
                   :: "v"(l0_), "v"(g0_) : "memory");                          \
      uint32_t l1_ = l0_ + 16u;                                                \
      uint64_t g1_ = g0_ + 16u;                                                \
      asm volatile("global_load_async_to_lds_b128 %0, %1, off"                 \
                   :: "v"(l1_), "v"(g1_) : "memory");                          \
    } while (0)

  ISSUE_PASS(0, 0);
  ISSUE_PASS(1, 1);

  float gv[8], bv[8];
  #pragma unroll
  for (int e = 0; e < 8; ++e) { gv[e] = lpg[sub * 8 + e]; bv[e] = lpb[sub * 8 + e]; }

  #pragma unroll 1
  for (int pass = 0; pass < 8; ++pass) {
    // retire this pass's 2 async instructions (next pass's 2 may remain)
#if __has_builtin(__builtin_amdgcn_s_wait_asynccnt)
    __builtin_amdgcn_s_wait_asynccnt(2);
#else
    asm volatile("s_wait_asynccnt 0x2" ::: "memory");
#endif
    int buf = pass & 1;
    int j = pass * 64 + rr;
    const float* st = &stage_s[buf][rr][sub * 8];
    float4 a0 = *(const float4*)(st);
    float4 a1 = *(const float4*)(st + 4);
    float v[8] = {a0.x, a0.y, a0.z, a0.w, a1.x, a1.y, a1.z, a1.w};
    float s = 0.f, sq = 0.f;
    #pragma unroll
    for (int e = 0; e < 8; ++e) { s += v[e]; sq += v[e] * v[e]; }
    #pragma unroll
    for (int o = 1; o < 8; o <<= 1) { s += __shfl_xor(s, o, 32); sq += __shfl_xor(sq, o, 32); }
    float mu = s * (1.0f / PP);
    float rs = rsqrtf(sq * (1.0f / PP) - mu * mu + LN_EPS);
    #pragma unroll
    for (int e = 0; e < 8; ++e)
      xp_s[j][sub * 8 + e] = f2bf((v[e] - mu) * rs * gv[e] + bv[e]);
    if (pass < 6) ISSUE_PASS(pass + 2, buf);
  }
  #undef ISSUE_PASS
  __syncthreads();

  // --- bias GEMM (WMMA): at_s[h][j] = sum_p xp[j][p] * Wb[h][p] ---
  for (int t = wave; t < 64; t += 16) {
    int jt = t & 31, ht = t >> 5;
    v8f acc = {0.f,0.f,0.f,0.f,0.f,0.f,0.f,0.f};
    #pragma unroll
    for (int kk = 0; kk < 2; ++kk) {
      v16bf a = load_frag_bf16(&xp_s[jt * 16][kk * 32], XP_LD);
      v16bf b = load_frag_bf16(Wb_bf + (size_t)ht * 16 * PP + kk * 32, PP);
      acc = __builtin_amdgcn_wmma_f32_16x16x32_bf16(false, a, false, b, (short)0, acc, false, false);
    }
    int nn = lane & 15, mb = (lane >> 4) << 3;
    int h = ht * 16 + nn, jb = jt * 16 + mb;
    #pragma unroll
    for (int r = 0; r < 8; ++r) at_s[h][jb + r] = acc[r];
  }
  __syncthreads();

  // --- logits: at_s[h][j] += q_i . k_j (kT L2-resident, coalesced over j) ---
  const float* kc = kT + (size_t)c * EE * NN;
  #pragma unroll 1
  for (int h = 0; h < HH; ++h) {
    float acc = at_s[h][tid];
    const float* kp = kc + (size_t)h * DD * NN + tid;
    #pragma unroll
    for (int d = 0; d < DD; ++d) acc += qrow_s[h * DD + d] * kp[(size_t)d * NN];
    at_s[h][tid] = acc;
  }
  __syncthreads();

  // --- softmax per head row (bb[h] cancels in softmax) ---
  for (int h = wave; h < HH; h += 16) {
    float m = -3.0e38f;
    for (int jj = lane; jj < NN; jj += 32) m = fmaxf(m, at_s[h][jj]);
    #pragma unroll
    for (int o = 16; o >= 1; o >>= 1) m = fmaxf(m, __shfl_xor(m, o, 32));
    float ssum = 0.f;
    for (int jj = lane; jj < NN; jj += 32) {
      float ev = __expf(at_s[h][jj] - m);
      at_s[h][jj] = ev;
      ssum += ev;
    }
    #pragma unroll
    for (int o = 16; o >= 1; o >>= 1) ssum += __shfl_xor(ssum, o, 32);
    if (lane == 0) inv_s[h] = 1.0f / ssum;
  }
  __syncthreads();

  // --- s[j] = sum_h probs[h][j] * pv[c][h][j]; force = sum_j s[j]*delta ---
  const float* pvc = pvT + (size_t)c * HH * NN;
  float sj = 0.f;
  #pragma unroll 1
  for (int h = 0; h < HH; ++h)
    sj += at_s[h][tid] * inv_s[h] * pvc[(size_t)h * NN + tid];

  float fx = sj * dp[0], fy = sj * dp[1], fz = sj * dp[2];
  #pragma unroll
  for (int o = 16; o >= 1; o >>= 1) {
    fx += __shfl_xor(fx, o, 32);
    fy += __shfl_xor(fy, o, 32);
    fz += __shfl_xor(fz, o, 32);
  }
  if (lane == 0) { red_s[wave][0] = fx; red_s[wave][1] = fy; red_s[wave][2] = fz; }
  __syncthreads();
  if (wave == 0) {
    float a  = (lane < 16) ? red_s[lane][0] : 0.f;
    float b  = (lane < 16) ? red_s[lane][1] : 0.f;
    float c2 = (lane < 16) ? red_s[lane][2] : 0.f;
    #pragma unroll
    for (int o = 8; o >= 1; o >>= 1) {
      a += __shfl_xor(a, o, 32);
      b += __shfl_xor(b, o, 32);
      c2 += __shfl_xor(c2, o, 32);
    }
    if (lane == 0) {
      out[(size_t)ci * 3 + 0] = a;
      out[(size_t)ci * 3 + 1] = b;
      out[(size_t)ci * 3 + 2] = c2;
    }
  }
}

extern "C" void kernel_launch(void* const* d_in, const int* in_sizes, int n_in,
                              void* d_out, int out_size, void* d_ws, size_t ws_size,
                              hipStream_t stream) {
  (void)in_sizes; (void)n_in; (void)out_size; (void)ws_size;
  const float* query  = (const float*)d_in[0];
  const float* pair   = (const float*)d_in[1];
  const float* delta  = (const float*)d_in[2];
  const float* ln_q_g = (const float*)d_in[3];
  const float* ln_q_b = (const float*)d_in[4];
  const float* ln_p_g = (const float*)d_in[5];
  const float* ln_p_b = (const float*)d_in[6];
  const float* Wq     = (const float*)d_in[7];
  const float* Wk     = (const float*)d_in[8];
  const float* Wv     = (const float*)d_in[9];
  const float* Wb     = (const float*)d_in[10];
  // d_in[11] = bb: constant along softmax axis -> cancels, unused.
  const float* Wf     = (const float*)d_in[12];
  float* out = (float*)d_out;

  char* ws = (char*)d_ws;
  bf16_t* xq    = (bf16_t*)ws;
  size_t o0 = (size_t)NC * NN * EE * sizeof(bf16_t);
  float* qbuf   = (float*)(ws + o0);
  size_t o1 = o0 + (size_t)NC * NN * EE * sizeof(float);
  float* kT     = (float*)(ws + o1);
  size_t o2 = o1 + (size_t)NC * EE * NN * sizeof(float);
  float* pvT    = (float*)(ws + o2);
  size_t o3 = o2 + (size_t)NC * HH * NN * sizeof(float);
  bf16_t* Wq_bf = (bf16_t*)(ws + o3);
  size_t o4 = o3 + (size_t)EE * EE * sizeof(bf16_t);
  bf16_t* Wk_bf = (bf16_t*)(ws + o4);
  size_t o5 = o4 + (size_t)EE * EE * sizeof(bf16_t);
  bf16_t* Wvf_bf = (bf16_t*)(ws + o5);
  size_t o6 = o5 + (size_t)HH * EE * sizeof(bf16_t);
  bf16_t* Wb_bf = (bf16_t*)(ws + o6);

  k_ln_query<<<NC * NN, 256, 0, stream>>>(query, ln_q_g, ln_q_b, xq);
  k_prep_weights<<<(EE * EE) / 256, 256, 0, stream>>>(Wq, Wk, Wv, Wb, Wf,
                                                      Wq_bf, Wk_bf, Wvf_bf, Wb_bf);
  k_qkv_gemm<<<dim3(128, 3), 256, 0, stream>>>(xq, Wq_bf, Wk_bf, Wvf_bf, qbuf, kT, pvT);
  k_attn<<<NC * NN, 512, 0, stream>>>(pair, delta, ln_p_g, ln_p_b, Wb_bf,
                                      qbuf, kT, pvT, out);
}
